// DETRsmpl_26001732010623
// MI455X (gfx1250) — matrix operand order repaired and checked by reading, not compiled, not run on hardware
//
#include <hip/hip_runtime.h>
#include <stdint.h>
#include <stddef.h>

// ---------------- problem constants ----------------
#define NPOSE   144
#define NBETA   10
#define NCAM    3
#define NTHETA  157      // NPOSE + NBETA + NCAM
#define CDIM    256      // feature width C, also hidden width
#define K1PAD   448      // 413 (=256+157) padded up to multiple of 64
#define KTPAD   192      // theta K (157) padded in LDS
#define N3PAD   160      // GEMM3 N (157) padded to multiple of 16
#define NITER   3
#define NROWS   (6 * 64 * 300)   // 115200

// ---------------- tiling ----------------
#define ROWS_PER_BLOCK 64        // 4 row-groups x 16 rows; 2 N-split waves each
#define THREADS        256       // 8 wave32
#define KP             64        // weight K-panel depth staged in LDS

// LDS element offsets (bf16 elements from smem base); total 128KB
#define ACT_E 0                  // 64 x 256 bf16 (32KB); sTheta overlays (64 x 192)
#define TH_E  0
#define SX_E  16384              // 64 x 256 bf16 (32KB) x staged as bf16
#define PAN_E 32768              // 2 x (256 x 64) bf16 panels (2 x 32KB)
#define PAN_STRIDE_E 16384
#define SMEM_BYTES 131072

typedef __attribute__((ext_vector_type(16))) __bf16 v16bf;
typedef __attribute__((ext_vector_type(8)))  __bf16 v8bf;
typedef __attribute__((ext_vector_type(4)))  __bf16 v4bf;
typedef __attribute__((ext_vector_type(8)))  float  v8f;
typedef __attribute__((ext_vector_type(4)))  float  v4f;
typedef __attribute__((ext_vector_type(4)))  unsigned int u32x4;
typedef __attribute__((ext_vector_type(8)))  int i32x8;
typedef __attribute__((ext_vector_type(4)))  int i32x4;

#if __has_builtin(__builtin_amdgcn_tensor_load_to_lds)
#define USE_TDM 1
#else
#define USE_TDM 0
#endif

// =====================================================================
// Kernel 0: convert fp32 weights -> bf16, transposed to Wt[N][K] with
// zero padding, stored contiguously in workspace.
//   W1t: [256][448]  (k<413 valid; k>=413 zero)
//   W2t: [256][256]
//   W3t: [160][256]  (n<157 valid; n>=157 zero)
// =====================================================================
__global__ void cvt_weights_kernel(const float* __restrict__ W1,
                                   const float* __restrict__ W2,
                                   const float* __restrict__ W3,
                                   __bf16* __restrict__ ws) {
  const int E1 = 256 * K1PAD;
  const int E2 = 256 * 256;
  const int E3 = N3PAD * 256;
  int idx = blockIdx.x * blockDim.x + threadIdx.x;
  if (idx < E1) {
    int n = idx / K1PAD, k = idx % K1PAD;
    float v = (k < 413) ? W1[(size_t)k * 256 + n] : 0.0f;
    ws[idx] = (__bf16)v;
  } else if (idx < E1 + E2) {
    int r = idx - E1;
    int n = r / 256, k = r % 256;
    ws[idx] = (__bf16)W2[(size_t)k * 256 + n];
  } else if (idx < E1 + E2 + E3) {
    int r = idx - E1 - E2;
    int n = r / 256, k = r % 256;
    float v = (n < NTHETA) ? W3[(size_t)k * NTHETA + n] : 0.0f;
    ws[idx] = (__bf16)v;
  }
}

// ---------------- fragment helpers ----------------
__device__ __forceinline__ v16bf cat8(v8bf a, v8bf b) {
  return __builtin_shufflevector(a, b, 0, 1, 2, 3, 4, 5, 6, 7,
                                       8, 9, 10, 11, 12, 13, 14, 15);
}

// A-fragment (16x32 bf16) from LDS: element offset `off` from smem base,
// lane holds K runs [off,off+8) and [off+16,off+24).
__device__ __forceinline__ v16bf afrag_lds(const __bf16* sb, int off) {
  v8bf lo = *(const v8bf*)(sb + off);
  v8bf hi = *(const v8bf*)(sb + off + 16);
  return cat8(lo, hi);
}

// One K=64 panel of WMMAs over NT 16-wide N tiles starting at tile TB.
// B-fragment loads for tile t+1 are software-pipelined over tile t's WMMAs.
template <int NT>
__device__ __forceinline__ void panel_mma(v8f* acc, const __bf16* sb, int panE,
                                          int TB, const v16bf* af,
                                          int laneN, int laneHi) {
  int bo = panE + (TB * 16 + laneN) * KP + laneHi * 8;
  v16bf b0 = cat8(*(const v8bf*)(sb + bo), *(const v8bf*)(sb + bo + 16));
  v16bf b1 = cat8(*(const v8bf*)(sb + bo + 32), *(const v8bf*)(sb + bo + 48));
#pragma unroll
  for (int t = 0; t < NT; ++t) {
    v16bf n0 = b0, n1 = b1;
    if (t + 1 < NT) {
      int no = bo + (t + 1) * (16 * KP);
      n0 = cat8(*(const v8bf*)(sb + no), *(const v8bf*)(sb + no + 16));
      n1 = cat8(*(const v8bf*)(sb + no + 32), *(const v8bf*)(sb + no + 48));
    }
    acc[t] = __builtin_amdgcn_wmma_f32_16x16x32_bf16(
        false, af[0], false, b0, (short)0, acc[t], false, false);
    acc[t] = __builtin_amdgcn_wmma_f32_16x16x32_bf16(
        false, af[1], false, b1, (short)0, acc[t], false, false);
    b0 = n0;
    b1 = n1;
  }
}

// Fallback cooperative panel copy (used only if TDM builtin is absent).
__device__ __forceinline__ void load_panel(__bf16* dst, const __bf16* Wt,
                                           int kpitch, int kp, int nrows, int tid) {
  const __bf16* src = Wt + kp * KP;
  int vecs = nrows * (KP / 8);
  for (int v = tid; v < vecs; v += THREADS) {
    int n  = v >> 3;
    int kk = (v & 7) << 3;
    *(uint4*)(dst + n * KP + kk) =
        *(const uint4*)(src + (size_t)n * kpitch + kk);
  }
}

#if USE_TDM
// ---------------------------------------------------------------------
// Tensor Data Mover: DMA a [rows][KP] bf16 tile (row pitch = kpitch
// elements) from global memory into LDS at byte address ldsOff.
// D# per CDNA5 ISA ch.8: group0 = {count, lds_addr, global_addr, type=2},
// group1 = {data_size=2B, tensor_dim0/1, tile_dim0/1, dim0_stride}.
// Groups 2/3 unused for 2-D tensors. 6-arg builtin form on this
// toolchain: (g0, g1, g2, g3, g_extra, cpol). Tracked by TENSORcnt.
// ---------------------------------------------------------------------
__device__ __forceinline__ void tdm_panel_load(unsigned ldsOff, const __bf16* gsrc,
                                               int rows, int kpitch) {
  unsigned long long ga = (unsigned long long)(size_t)gsrc;
  u32x4 g0;
  g0[0] = 1u;                                   // count=1, user descriptor
  g0[1] = ldsOff;                               // lds_addr (bytes)
  g0[2] = (unsigned)(ga & 0xffffffffu);         // global_addr[31:0]
  g0[3] = (unsigned)((ga >> 32) & 0x1ffffffu)   // global_addr[56:32]
        | (2u << 30);                           // type = 2 ("image")

  unsigned d0 = (unsigned)KP;                   // tensor_dim0 (elements)
  unsigned d1 = (unsigned)rows;                 // tensor_dim1
  unsigned long long s0 = (unsigned long long)kpitch;  // dim0 stride (elements)
  i32x8 g1;
  g1[0] = (int)(1u << 16);                      // data_size=1 (2 bytes), wg_mask=0
  g1[1] = (int)((d0 & 0xffffu) << 16);          // tensor_dim0[15:0] @ bits 63:48
  g1[2] = (int)((d0 >> 16) | ((d1 & 0xffffu) << 16));
  g1[3] = (int)((d1 >> 16) | ((unsigned)KP << 16));   // tile_dim0 = KP
  g1[4] = (int)(d1 & 0xffffu);                  // tile_dim1 = rows, tile_dim2 = 0
  g1[5] = (int)(s0 & 0xffffffffu);              // tensor_dim0_stride[31:0]
  g1[6] = (int)((s0 >> 32) & 0xffffu);          // tensor_dim0_stride[47:32]
  g1[7] = 0;                                    // tensor_dim1_stride = 0 (2D)

  i32x4 z4 = {0, 0, 0, 0};
  i32x8 z8 = {0, 0, 0, 0, 0, 0, 0, 0};
  __builtin_amdgcn_tensor_load_to_lds(g0, g1, z4, z4, z8, 0);
}

#define PANEL_PREFETCH(Wt, kpitch, rows)                                    \
  do {                                                                      \
    if (wave == 0) tdm_panel_load(panByte0, (Wt), (rows), (kpitch));        \
  } while (0)

// Stage panel kp: issue DMA for kp+1 into the other buffer, then wait
// until only that new DMA is still in flight (double-buffered pipeline).
#define PANEL_STAGE(Wt, kpitch, rows, NP, kp)                               \
  do {                                                                      \
    if (wave == 0) {                                                        \
      if ((kp) + 1 < (NP)) {                                                \
        tdm_panel_load((((kp) & 1) ? panByte0 : panByte1),                  \
                       (Wt) + (size_t)((kp) + 1) * KP, (rows), (kpitch));   \
        __builtin_amdgcn_s_wait_tensorcnt(1);                               \
      } else {                                                              \
        __builtin_amdgcn_s_wait_tensorcnt(0);                               \
      }                                                                     \
    }                                                                       \
  } while (0)
#else
#define PANEL_PREFETCH(Wt, kpitch, rows) do { } while (0)
#define PANEL_STAGE(Wt, kpitch, rows, NP, kp)                               \
  load_panel(sw + PAN_E + ((kp) & 1) * PAN_STRIDE_E, (Wt), (kpitch), (kp),  \
             (rows), tid)
#endif

// =====================================================================
// Kernel 1: fully fused 3-iteration regression head + rot6d epilogue.
// 256 threads (8 wave32), 64 rows/block: wave = (row-group g = wave>>1,
// N-split s = wave&1). Each wave owns 16 rows x 128 cols -> acc[8]
// (64 VGPRs) and 5 theta tiles (40 VGPRs): ~half the registers of the
// unsplit version, so the 2-blocks/WGP LDS occupancy is VGPR-feasible.
// x is staged to LDS as bf16 once; weights stream L2 -> LDS via TDM
// double-buffered K-panels; all matmuls are v_wmma_f32_16x16x32_bf16.
// All LDS addressing is a single base + integer element offsets so the
// compiler keeps ds_load/ds_store (no generic flat fallback).
// =====================================================================
__global__ __launch_bounds__(THREADS, 1)
void hmr_head_kernel(const float* __restrict__ x,
                     const __bf16* __restrict__ wsW,
                     const float* __restrict__ b1,
                     const float* __restrict__ b2,
                     const float* __restrict__ b3,
                     const float* __restrict__ initp,
                     const float* __restrict__ inits,
                     const float* __restrict__ initc,
                     float* __restrict__ outR,
                     float* __restrict__ outB,
                     float* __restrict__ outC) {
  extern __shared__ char smem[];
  const __bf16* sb = (const __bf16*)smem;   // LDS read base
  __bf16*       sw = (__bf16*)smem;         // LDS write base
  float*        sposef = (float*)smem;      // 64 x 144 f32 overlay, final phase

  const int tid     = threadIdx.x;
  const int lane    = tid & 31;
  const int wave    = tid >> 5;
  const int laneN   = lane & 15;
  const int laneHi  = lane >> 4;
  const int g       = wave >> 1;            // row group 0..3
  const int s       = wave & 1;             // N split 0..1
  const int rowW    = g * 16;
  const int rowBase = blockIdx.x * ROWS_PER_BLOCK;
  const int TB12    = s * 8;                // first N tile, GEMM1/2 (8 tiles)
  const int TB3     = s * 5;                // first N tile, GEMM3 (5 tiles)

  const __bf16* W1t = wsW;
  const __bf16* W2t = W1t + 256 * K1PAD;
  const __bf16* W3t = W2t + 256 * 256;

  const unsigned smemA   = (unsigned)(size_t)(void*)smem;
  const unsigned panByte0 = smemA + 2u * PAN_E;
  const unsigned panByte1 = smemA + 2u * (PAN_E + PAN_STRIDE_E);
  (void)panByte0; (void)panByte1;

  // ---- stage x rows -> bf16 LDS (once per block) ----
  for (int v = tid; v < (ROWS_PER_BLOCK * CDIM) / 4; v += THREADS) {
    int r = v >> 6;                 // 64 float4 per row
    int c = (v & 63) << 2;
    v4f f = *(const v4f*)(x + (size_t)(rowBase + r) * CDIM + c);
    v4bf h;
#pragma unroll
    for (int i = 0; i < 4; ++i) h[i] = (__bf16)f[i];
    *(v4bf*)(sw + SX_E + r * CDIM + c) = h;
  }

  // theta master copy (this wave's 5 tiles), fp32, WMMA C/D layout:
  // lane holds col N = (TB3+j)*16+laneN, rows M = i + 8*laneHi
  v8f thC[5];
#pragma unroll
  for (int j = 0; j < 5; ++j) {
    int N = (TB3 + j) * 16 + laneN;
    float v;
    if (N < NPOSE)              v = initp[N];
    else if (N < NPOSE + NBETA) v = inits[N - NPOSE];
    else if (N < NTHETA)        v = initc[N - NPOSE - NBETA];
    else                        v = 0.0f;
#pragma unroll
    for (int i = 0; i < 8; ++i) thC[j][i] = v;
  }

  const int arowX = SX_E + (rowW + laneN) * CDIM;    // x bf16 row (A loads)
  const int arowT = TH_E + (rowW + laneN) * KTPAD;   // theta bf16 row
  const int arowA = ACT_E + (rowW + laneN) * CDIM;   // activations row

  v8f acc[8];

#pragma unroll 1
  for (int it = 0; it < NITER; ++it) {
    // theta regs -> bf16 row-major LDS (rows of this group, cols of this
    // split); s==0 waves also re-zero pad cols [160,192)
#pragma unroll
    for (int j = 0; j < 5; ++j) {
#pragma unroll
      for (int i = 0; i < 8; ++i) {
        sw[TH_E + (rowW + i + 8 * laneHi) * KTPAD + (TB3 + j) * 16 + laneN] =
            (__bf16)thC[j][i];
      }
    }
    if (s == 0) {
      v8bf z = {};
      int zo = TH_E + (rowW + laneN) * KTPAD + 160 + laneHi * 16;
      *(v8bf*)(sw + zo) = z;
      *(v8bf*)(sw + zo + 8) = z;
    }
    __syncthreads();

    // ---- GEMM1: h1 = relu([x, theta] @ W1 + b1), N=256, K=448 ----
#pragma unroll
    for (int t = 0; t < 8; ++t) {
      float bv = b1[(TB12 + t) * 16 + laneN];
#pragma unroll
      for (int i = 0; i < 8; ++i) acc[t][i] = bv;
    }
    PANEL_PREFETCH(W1t, K1PAD, 256);
#pragma unroll 1
    for (int kp = 0; kp < K1PAD / KP; ++kp) {
      int panE = PAN_E + (kp & 1) * PAN_STRIDE_E;
      PANEL_STAGE(W1t, K1PAD, 256, K1PAD / KP, kp);
      __syncthreads();
      int kg = kp * KP;
      v16bf af[2];
#pragma unroll
      for (int kk = 0; kk < 2; ++kk) {
        int kb = kg + kk * 32 + laneHi * 8;
        int ao = (kg < CDIM) ? (arowX + kb) : (arowT + kb - CDIM);
        af[kk] = afrag_lds(sb, ao);
      }
      panel_mma<8>(acc, sb, panE, TB12, af, laneN, laneHi);
      __syncthreads();
    }
    // relu, h1 -> sAct (bf16, row-major)
#pragma unroll
    for (int t = 0; t < 8; ++t) {
#pragma unroll
      for (int i = 0; i < 8; ++i) {
        float h = fmaxf(acc[t][i], 0.0f);
        sw[ACT_E + (rowW + i + 8 * laneHi) * CDIM + (TB12 + t) * 16 + laneN] =
            (__bf16)h;
      }
    }

    // ---- GEMM2: h2 = h1 @ W2 + b2, N=256, K=256 ----
#pragma unroll
    for (int t = 0; t < 8; ++t) {
      float bv = b2[(TB12 + t) * 16 + laneN];
#pragma unroll
      for (int i = 0; i < 8; ++i) acc[t][i] = bv;
    }
    PANEL_PREFETCH(W2t, CDIM, 256);
#pragma unroll 1
    for (int kp = 0; kp < CDIM / KP; ++kp) {
      int panE = PAN_E + (kp & 1) * PAN_STRIDE_E;
      PANEL_STAGE(W2t, CDIM, 256, CDIM / KP, kp);
      __syncthreads();   // also publishes h1 on the first panel
      v16bf af[2];
#pragma unroll
      for (int kk = 0; kk < 2; ++kk) {
        af[kk] = afrag_lds(sb, arowA + kp * KP + kk * 32 + laneHi * 8);
      }
      panel_mma<8>(acc, sb, panE, TB12, af, laneN, laneHi);
      __syncthreads();
    }
    // h2 -> sAct (all h1 reads completed at the final barrier above)
#pragma unroll
    for (int t = 0; t < 8; ++t) {
#pragma unroll
      for (int i = 0; i < 8; ++i) {
        sw[ACT_E + (rowW + i + 8 * laneHi) * CDIM + (TB12 + t) * 16 + laneN] =
            (__bf16)acc[t][i];
      }
    }

    // ---- GEMM3: theta += h2 @ W3 + b3, N=160(157), K=256 ----
#pragma unroll
    for (int j = 0; j < 5; ++j) {
      int N = (TB3 + j) * 16 + laneN;
      float bv = (N < NTHETA) ? b3[N] : 0.0f;
#pragma unroll
      for (int i = 0; i < 8; ++i) acc[j][i] = bv;
    }
    PANEL_PREFETCH(W3t, CDIM, N3PAD);
#pragma unroll 1
    for (int kp = 0; kp < CDIM / KP; ++kp) {
      int panE = PAN_E + (kp & 1) * PAN_STRIDE_E;
      PANEL_STAGE(W3t, CDIM, N3PAD, CDIM / KP, kp);
      __syncthreads();   // also publishes h2 on the first panel
      v16bf af[2];
#pragma unroll
      for (int kk = 0; kk < 2; ++kk) {
        af[kk] = afrag_lds(sb, arowA + kp * KP + kk * 32 + laneHi * 8);
      }
      panel_mma<5>(acc, sb, panE, TB3, af, laneN, laneHi);
      __syncthreads();
    }
#pragma unroll
    for (int j = 0; j < 5; ++j) {
#pragma unroll
      for (int i = 0; i < 8; ++i) thC[j][i] += acc[j][i];
    }
  }

  // ---- epilogue: betas & camera from registers (tile 9 lives in s==1, j==4)
  if (s == 1) {
#pragma unroll
    for (int i = 0; i < 8; ++i) {
      int gr  = rowBase + rowW + i + 8 * laneHi;
      float v = thC[4][i];
      if (laneN < NBETA) {
        outB[(size_t)gr * NBETA + laneN] = v;
      } else if (laneN < NBETA + NCAM) {
        outC[(size_t)gr * NCAM + (laneN - NBETA)] = v;
      }
    }
  }

  __syncthreads();
  // pose cols (N<144) -> fp32 LDS, row-major (s==0: tiles 0..4, s==1: 5..8)
  if (s == 0) {
#pragma unroll
    for (int j = 0; j < 5; ++j) {
#pragma unroll
      for (int i = 0; i < 8; ++i) {
        sposef[(rowW + i + 8 * laneHi) * NPOSE + j * 16 + laneN] = thC[j][i];
      }
    }
  } else {
#pragma unroll
    for (int j = 0; j < 4; ++j) {
#pragma unroll
      for (int i = 0; i < 8; ++i) {
        sposef[(rowW + i + 8 * laneHi) * NPOSE + 80 + j * 16 + laneN] = thC[j][i];
      }
    }
  }
  __syncthreads();

  // rot6d -> 3x3 rotation matrices (Gram-Schmidt), 24 joints per row
  for (int task = tid; task < ROWS_PER_BLOCK * 24; task += THREADS) {
    int r = task / 24, j = task % 24;
    const float* p = sposef + r * NPOSE + j * 6;
    float a1x = p[0], a2x = p[1];
    float a1y = p[2], a2y = p[3];
    float a1z = p[4], a2z = p[5];
    float inv1 = 1.0f / fmaxf(sqrtf(a1x * a1x + a1y * a1y + a1z * a1z), 1e-12f);
    float b1x = a1x * inv1, b1y = a1y * inv1, b1z = a1z * inv1;
    float d   = b1x * a2x + b1y * a2y + b1z * a2z;
    float cx  = a2x - d * b1x, cy = a2y - d * b1y, cz = a2z - d * b1z;
    float inv2 = 1.0f / fmaxf(sqrtf(cx * cx + cy * cy + cz * cz), 1e-12f);
    float b2x = cx * inv2, b2y = cy * inv2, b2z = cz * inv2;
    float b3x = b1y * b2z - b1z * b2y;
    float b3y = b1z * b2x - b1x * b2z;
    float b3z = b1x * b2y - b1y * b2x;
    float* o = outR + ((size_t)(rowBase + r) * 24 + j) * 9;
    o[0] = b1x; o[1] = b2x; o[2] = b3x;
    o[3] = b1y; o[4] = b2y; o[5] = b3y;
    o[6] = b1z; o[7] = b2z; o[8] = b3z;
  }
}

// =====================================================================
extern "C" void kernel_launch(void* const* d_in, const int* in_sizes, int n_in,
                              void* d_out, int out_size, void* d_ws, size_t ws_size,
                              hipStream_t stream) {
  (void)in_sizes; (void)n_in; (void)out_size; (void)ws_size;
  const float* x     = (const float*)d_in[0];
  // d_in[1] = pred_class (unused by the reference computation)
  const float* W1    = (const float*)d_in[2];
  const float* b1    = (const float*)d_in[3];
  const float* W2    = (const float*)d_in[4];
  const float* b2    = (const float*)d_in[5];
  const float* W3    = (const float*)d_in[6];
  const float* b3    = (const float*)d_in[7];
  const float* initp = (const float*)d_in[8];
  const float* inits = (const float*)d_in[9];
  const float* initc = (const float*)d_in[10];

  __bf16* ws = (__bf16*)d_ws;

  float* outR = (float*)d_out;                       // NROWS * 24*3*3
  float* outB = outR + (size_t)NROWS * 216;          // NROWS * 10
  float* outC = outB + (size_t)NROWS * NBETA;        // NROWS * 3

  const int cvtTotal = 256 * K1PAD + 256 * 256 + N3PAD * 256;
  cvt_weights_kernel<<<(cvtTotal + 255) / 256, 256, 0, stream>>>(W1, W2, W3, ws);

  dim3 grid(NROWS / ROWS_PER_BLOCK);   // 1800
  hmr_head_kernel<<<grid, THREADS, SMEM_BYTES, stream>>>(
      x, ws, b1, b2, b3, initp, inits, initc, outR, outB, outC);
}
